// LiteMLA_60739427500919
// MI455X (gfx1250) — matrix operand order repaired and checked by reading, not compiled, not run on hardware
//
#include <hip/hip_runtime.h>

typedef __attribute__((ext_vector_type(16))) _Float16 v16h;
typedef __attribute__((ext_vector_type(8)))  _Float16 v8h;
typedef __attribute__((ext_vector_type(8)))  float    v8f;
typedef __attribute__((ext_vector_type(4)))  float    f4;

#define HW 4096
#define IMGW 64

// ---------------------------------------------------------------------------
// WMMA GEMM for 1x1 convs in NCHW:
//   Out[b, c, hw] = bias[c] + sum_k W[c, k] * In[b, k, hw]
// Block: 256 threads (8 wave32), tile = 64 channels x 64 spatial, K-step 32.
// LDS tiles live in WMMA-fragment order:
//   A frag (16x32 f16): lanes 0-15 row=lane,  K pos order {0..7,16..23}
//                       lanes 16-31 row=lane-16, K {8..15,24..31}
//   B frag (32x16 f16): lanes 0-15 K {0..7,16..23}, lanes 16-31 K {8..15,24..31}, col=lane&15
// Staging writes one contiguous 8-half run per thread (1x ds_store_b128 per
// tile per thread); all global pointers are loop-carried with immediate
// offsets; all LDS addresses are loop-invariant.
// Each wave holds two M-subtiles sharing one B fragment: 2 wmma per B read.
// ---------------------------------------------------------------------------
template <int CIN, int COUT, typename TB>
__global__ __launch_bounds__(256) void gemm_wmma(const float* __restrict__ Wq,
                                                 const float* __restrict__ bias,
                                                 const TB* __restrict__ Bsrc,
                                                 float* __restrict__ Out) {
  // [hilo][row/col (64)][24 halves]; 48B row stride keeps fragment bases
  // 16B-aligned and spreads 16-lane b128 accesses across all 64 banks.
  __shared__ __align__(16) _Float16 As[2][64][24];
  __shared__ __align__(16) _Float16 Bs[2][64][24];

  const int t = threadIdx.x;
  const int lane = t & 31;
  const int wave = t >> 5;
  const int nc = wave & 3;    // spatial 16-subtile 0..3
  const int ms0 = wave >> 2;  // channel 16-subtiles {ms0, ms0+2}
  const int ms1 = ms0 + 2;

  const int m0 = blockIdx.x * 64;  // spatial tile start (0..32767)
  const int c0 = blockIdx.y * 64;  // channel tile start
  const int b = m0 / HW;
  const int hw0 = m0 % HW;

  const int hl = lane >> 4;
  const int l15 = lane & 15;

  // --- A staging assignment: row ar, K-octet akq (8 consecutive K) ---
  const int ar = t >> 2;
  const int akq = (t & 3) * 8;
  const int ahl = t & 1;                 // K-octet 0,16 -> 0 ; 8,24 -> 1
  const int aposb = ((t >> 1) & 1) * 8;  // K-octet 0,8 -> 0 ; 16,24 -> 8
  const float* wp = Wq + (size_t)(c0 + ar) * CIN + akq;
  _Float16* asl = &As[ahl][ar][aposb];

  // --- B staging assignment: col bcol, K-octet (t>>6)*8 ---
  const int bcol = t & 63;
  const int bk8 = (t >> 6) * 8;
  const int bhl = (t >> 6) & 1;          // K-octet 0,16 -> 0 ; 8,24 -> 1
  const int bposb = ((t >> 7) & 1) * 8;  // K-octet 0,8 -> 0 ; 16,24 -> 8
  const TB* bp = Bsrc + ((size_t)(b * CIN) + bk8) * HW + hw0 + bcol;
  _Float16* bsl = &Bs[bhl][bcol][bposb];

  // --- loop-invariant fragment pointers ---
  const _Float16* fb = &Bs[hl][nc * 16 + l15][0];
  const _Float16* fa0 = &As[hl][ms0 * 16 + l15][0];
  const _Float16* fa1 = &As[hl][ms1 * 16 + l15][0];

  v8f acc0 = {}, acc1 = {};

#pragma unroll 4
  for (int k0 = 0; k0 < CIN; k0 += 32) {
    if (k0 + 32 < CIN) __builtin_prefetch(bp + (size_t)32 * HW, 0, 2);

    // Stage A: 8 consecutive K of one row -> 2x global b128, 1x ds_store_b128
    f4 w0, w1;
    __builtin_memcpy(&w0, wp, 16);
    __builtin_memcpy(&w1, wp + 4, 16);
    v8h apk;
    apk[0] = (_Float16)w0[0]; apk[1] = (_Float16)w0[1];
    apk[2] = (_Float16)w0[2]; apk[3] = (_Float16)w0[3];
    apk[4] = (_Float16)w1[0]; apk[5] = (_Float16)w1[1];
    apk[6] = (_Float16)w1[2]; apk[7] = (_Float16)w1[3];
    __builtin_memcpy(asl, &apk, 16);

    // Stage B: 8 K of one column -> 8 coalesced b32 loads, 1x ds_store_b128
    v8h bpk;
#pragma unroll
    for (int j = 0; j < 8; ++j) bpk[j] = (_Float16)(float)bp[(size_t)j * HW];
    __builtin_memcpy(bsl, &bpk, 16);

    __syncthreads();

    v16h bfrag, a0, a1;
    __builtin_memcpy(&bfrag, fb, sizeof(v16h));
    __builtin_memcpy(&a0, fa0, sizeof(v16h));
    __builtin_memcpy(&a1, fa1, sizeof(v16h));

    acc0 = __builtin_amdgcn_wmma_f32_16x16x32_f16(false, a0, false, bfrag,
                                                  (short)0, acc0, false, false);
    acc1 = __builtin_amdgcn_wmma_f32_16x16x32_f16(false, a1, false, bfrag,
                                                  (short)0, acc1, false, false);
    __syncthreads();

    wp += 32;
    bp += (size_t)32 * HW;
  }

  // D layout: VGPR r -> row r (lanes 0-15) / row r+8 (lanes 16-31), col=lane&15
  const int hw = hw0 + nc * 16 + l15;
#pragma unroll
  for (int r = 0; r < 8; ++r) {
    int rr = r + ((lane < 16) ? 0 : 8);
    int ch0 = c0 + ms0 * 16 + rr;
    int ch1 = c0 + ms1 * 16 + rr;
    Out[(size_t)(b * COUT + ch0) * HW + hw] = acc0[r] + bias[ch0];
    Out[(size_t)(b * COUT + ch1) * HW + hw] = acc1[r] + bias[ch1];
  }
}

// ---------------------------------------------------------------------------
// Fused depthwise (2R+1)^2 conv + grouped 1x1 conv (24 in -> 24 out).
// Block: 256 threads = one 16x16 spatial tile for one group of 24 channels.
// Depthwise result lives only in LDS (never hits HBM).
// ---------------------------------------------------------------------------
template <int R>
__global__ __launch_bounds__(256) void dwpw_kernel(
    const float* __restrict__ qkv, const float* __restrict__ dww,
    const float* __restrict__ dwb, const float* __restrict__ pww,
    const float* __restrict__ pwb, _Float16* __restrict__ outp) {
  __shared__ float tile[24][256];
  __shared__ float wpw[24][24];
  __shared__ float bpw[24];

  const int t = threadIdx.x;
  const int g = blockIdx.y;       // group 0..31
  const int b = blockIdx.z;       // batch 0..7
  const int tileId = blockIdx.x;  // 0..15
  const int x = (tileId & 3) * 16 + (t & 15);
  const int y = (tileId >> 2) * 16 + (t >> 4);

  if (t < 24) bpw[t] = pwb[g * 24 + t];
  for (int i = t; i < 24 * 24; i += 256)
    wpw[i / 24][i % 24] = pww[(g * 24 + i / 24) * 24 + (i % 24)];

  const int K = 2 * R + 1;
  for (int c = 0; c < 24; ++c) {
    const int ch = g * 24 + c;
    const float* src = qkv + (size_t)(b * 768 + ch) * HW;
    float a = dwb[ch];
#pragma unroll
    for (int dy = -R; dy <= R; ++dy) {
      int yy = y + dy;
      if (yy < 0 || yy >= IMGW) continue;
#pragma unroll
      for (int dx = -R; dx <= R; ++dx) {
        int xx = x + dx;
        if (xx < 0 || xx >= IMGW) continue;
        a += dww[ch * K * K + (dy + R) * K + (dx + R)] * src[yy * IMGW + xx];
      }
    }
    tile[c][t] = a;
  }
  __syncthreads();

#pragma unroll 4
  for (int oc = 0; oc < 24; ++oc) {
    float a = bpw[oc];
#pragma unroll
    for (int ic = 0; ic < 24; ++ic) a += wpw[oc][ic] * tile[ic][t];
    outp[(size_t)(b * 768 + g * 24 + oc) * HW + y * IMGW + x] = (_Float16)a;
  }
}

// ---------------------------------------------------------------------------
// ReLU linear attention. multi = [qkv(f32) | s3(f16) | s5(f16)], 2304 chans,
// 96 heads of 24 rows (q:0-7 relu, k:8-15 relu, v:16-23 + ones row).
// One block per (head, batch): vk[9][8] reduction then rank-9 replay.
// ---------------------------------------------------------------------------
__device__ __forceinline__ float fetch_multi(const float* qkv,
                                             const _Float16* s3,
                                             const _Float16* s5, int b,
                                             int head, int r, int n) {
  const int src = head >> 5;
  const int ch = (head & 31) * 24 + r;
  const size_t idx = (size_t)(b * 768 + ch) * HW + n;
  if (src == 0) return qkv[idx];
  if (src == 1) return (float)s3[idx];
  return (float)s5[idx];
}

__global__ __launch_bounds__(256) void attn_kernel(
    const float* __restrict__ qkv, const _Float16* __restrict__ s3,
    const _Float16* __restrict__ s5, _Float16* __restrict__ att) {
  __shared__ float vk[9][8];
  const int t = threadIdx.x;
  const int lane = t & 31;
  const int head = blockIdx.x;
  const int b = blockIdx.y;

  if (t < 72) ((float*)vk)[t] = 0.f;
  __syncthreads();

  float p[9][8];
#pragma unroll
  for (int d = 0; d < 9; ++d)
#pragma unroll
    for (int e = 0; e < 8; ++e) p[d][e] = 0.f;

  for (int n = t; n < HW; n += 256) {
    float kv[8], vv[8];
#pragma unroll
    for (int e = 0; e < 8; ++e) {
      float kk = fetch_multi(qkv, s3, s5, b, head, 8 + e, n);
      kv[e] = kk > 0.f ? kk : 0.f;
    }
#pragma unroll
    for (int d = 0; d < 8; ++d) vv[d] = fetch_multi(qkv, s3, s5, b, head, 16 + d, n);
#pragma unroll
    for (int d = 0; d < 8; ++d)
#pragma unroll
      for (int e = 0; e < 8; ++e) p[d][e] += vv[d] * kv[e];
#pragma unroll
    for (int e = 0; e < 8; ++e) p[8][e] += kv[e];  // ones-row of v
  }

  // wave-level tree reduce, then one LDS atomic per wave per element
#pragma unroll
  for (int d = 0; d < 9; ++d)
#pragma unroll
    for (int e = 0; e < 8; ++e) {
      float v = p[d][e];
#pragma unroll
      for (int off = 16; off > 0; off >>= 1) v += __shfl_down(v, off, 32);
      if (lane == 0) atomicAdd(&vk[d][e], v);
    }
  __syncthreads();

  for (int n = t; n < HW; n += 256) {
    float q[8];
#pragma unroll
    for (int e = 0; e < 8; ++e) {
      float qq = fetch_multi(qkv, s3, s5, b, head, e, n);
      q[e] = qq > 0.f ? qq : 0.f;
    }
    float o[9];
#pragma unroll
    for (int d = 0; d < 9; ++d) {
      float a = 0.f;
#pragma unroll
      for (int e = 0; e < 8; ++e) a += vk[d][e] * q[e];
      o[d] = a;
    }
    const float inv = 1.f / (o[8] + 1e-15f);
#pragma unroll
    for (int d = 0; d < 8; ++d)
      att[(size_t)(b * 768 + head * 8 + d) * HW + n] = (_Float16)(o[d] * inv);
  }
}

// ---------------------------------------------------------------------------
extern "C" void kernel_launch(void* const* d_in, const int* in_sizes, int n_in,
                              void* d_out, int out_size, void* d_ws,
                              size_t ws_size, hipStream_t stream) {
  const float* x = (const float*)d_in[0];
  const float* qkv_w = (const float*)d_in[1];
  const float* qkv_b = (const float*)d_in[2];
  const float* dw3_w = (const float*)d_in[3];
  const float* dw3_b = (const float*)d_in[4];
  const float* pw3_w = (const float*)d_in[5];
  const float* pw3_b = (const float*)d_in[6];
  const float* dw5_w = (const float*)d_in[7];
  const float* dw5_b = (const float*)d_in[8];
  const float* pw5_w = (const float*)d_in[9];
  const float* pw5_b = (const float*)d_in[10];
  const float* proj_w = (const float*)d_in[11];
  const float* proj_b = (const float*)d_in[12];

  const size_t E = (size_t)8 * 768 * HW;  // 25,165,824 elements
  float* qkv = (float*)d_ws;              // fp32: needed by dw convs + attn
  _Float16* s3 = (_Float16*)(qkv + E);    // fp16 intermediates: halve traffic
  _Float16* s5 = s3 + E;
  _Float16* att = s5 + E;

  // qkv = W(768x256) @ x  — WMMA GEMM, grid: 512 spatial x 12 channel tiles
  gemm_wmma<256, 768, float>
      <<<dim3(512, 12), 256, 0, stream>>>(qkv_w, qkv_b, x, qkv);

  // s3 = pw3(dw3(qkv)), s5 = pw5(dw5(qkv)) — fused, dw result stays in LDS
  dwpw_kernel<1>
      <<<dim3(16, 32, 8), 256, 0, stream>>>(qkv, dw3_w, dw3_b, pw3_w, pw3_b, s3);
  dwpw_kernel<2>
      <<<dim3(16, 32, 8), 256, 0, stream>>>(qkv, dw5_w, dw5_b, pw5_w, pw5_b, s5);

  // att = relu-linear-attention over concat([qkv, s3, s5])
  attn_kernel<<<dim3(96, 8), 256, 0, stream>>>(qkv, s3, s5, att);

  // out = proj(768->256) — WMMA GEMM, fp16 activations in, fp32 out
  gemm_wmma<768, 256, _Float16>
      <<<dim3(512, 4), 256, 0, stream>>>(proj_w, proj_b, att, (float*)d_out);
}
